// DGCNN_18657337933843
// MI455X (gfx1250) — matrix operand order repaired and checked by reading, not compiled, not run on hardware
//
#include <hip/hip_runtime.h>
#include <hip/hip_bf16.h>

#define N_      32768
#define H_      128
#define E_      524288
#define B_      64
#define NPG_    512
#define K_      30
#define LAYERS_ 4

typedef __attribute__((ext_vector_type(16))) __bf16 v16bf;
typedef __attribute__((ext_vector_type(8)))  __bf16 v8bf;
typedef __attribute__((ext_vector_type(8)))  float  v8f;

// ---------------------------------------------------------------- convert f32 -> bf16
__global__ __launch_bounds__(256) void cvt_bf16_kernel(const float* __restrict__ x,
                                                       __bf16* __restrict__ xb, int n) {
    int t = blockIdx.x * 256 + threadIdx.x;
    if (t < n) xb[t] = (__bf16)x[t];
}

// ---------------------------------------------------------------- pack W [128x128] into WMMA B-operand layout
// wpack layout: [kb(4)][nb(8)][lane(32)][16 bf16]; lane<16: N=nb*16+lane, K=kb*32+(0..15)
//                                                  lane>=16: N=nb*16+lane-16, K=kb*32+(16..31)
__global__ __launch_bounds__(256) void pack_w_kernel(const float* __restrict__ W,
                                                     __bf16* __restrict__ wp) {
    int t = blockIdx.x * 256 + threadIdx.x;      // 0 .. 16383
    if (t >= 16384) return;
    int i    = t & 15;
    int lane = (t >> 4) & 31;
    int nb   = (t >> 9) & 7;
    int kb   = t >> 12;
    int n  = nb * 16 + (lane & 15);
    int kl = ((lane >> 4) << 4) + i;             // 0..15 or 16..31
    int k  = kb * 32 + kl;
    wp[t] = (__bf16)W[k * H_ + n];
}

// ---------------------------------------------------------------- xw = x @ W  via v_wmma_f32_16x16x32_bf16
// 256 threads = 8 waves; each wave: one 16-row block x all 128 cols (8 nb tiles), K=128 in 4 steps.
// Per k-step: batch-load all 8 B fragments (one ds clause, one wait), then 8 back-to-back WMMAs.
__global__ __launch_bounds__(256) void gcn_gemm_kernel(const __bf16* __restrict__ xb,
                                                       const __bf16* __restrict__ wpack,
                                                       float* __restrict__ xw) {
    __shared__ __attribute__((aligned(32))) __bf16 wl[16384];   // 32 KB
    {   // cooperative stage of packed W
        const uint4* s = (const uint4*)wpack;
        uint4*       d = (uint4*)wl;
        for (int i = threadIdx.x; i < 2048; i += 256) d[i] = s[i];
    }
    __syncthreads();

    const int wave = threadIdx.x >> 5;
    const int lane = threadIdx.x & 31;
    const int rowblk = blockIdx.x * 8 + wave;           // 16-row block id
    const int rbase  = rowblk * 16;
    const int mrow   = lane & 15;
    const __bf16* arow = xb + (size_t)(rbase + mrow) * H_;
    const int off0 = (lane < 16) ? 0 : 16;              // A layout: one 16-elem contiguous K run per lane

    v8f acc[8];
    const v8f z = {0.f,0.f,0.f,0.f,0.f,0.f,0.f,0.f};
    #pragma unroll
    for (int nb = 0; nb < 8; ++nb) acc[nb] = z;

    #pragma unroll
    for (int kb = 0; kb < 4; ++kb) {
        v8bf lo = *(const v8bf*)(arow + kb * 32 + off0);
        v8bf hi = *(const v8bf*)(arow + kb * 32 + off0 + 8);
        v16bf a = __builtin_shufflevector(lo, hi, 0,1,2,3,4,5,6,7,8,9,10,11,12,13,14,15);

        // batch all B fragments for this k-step into registers first
        v16bf bfrag[8];
        #pragma unroll
        for (int nb = 0; nb < 8; ++nb)
            bfrag[nb] = *(const v16bf*)&wl[(((kb * 8) + nb) * 32 + lane) * 16];

        #pragma unroll
        for (int nb = 0; nb < 8; ++nb)
            acc[nb] = __builtin_amdgcn_wmma_f32_16x16x32_bf16(
                false, a, false, bfrag[nb], (short)0, acc[nb], false, false);
    }

    // C/D layout: VGPR r -> M = r (lanes 0-15) or 8+r (lanes 16-31); N = lane&15
    const int rofs = (lane < 16) ? 0 : 8;
    const int ncol = lane & 15;
    #pragma unroll
    for (int nb = 0; nb < 8; ++nb)
        #pragma unroll
        for (int r = 0; r < 8; ++r)
            xw[(size_t)(rbase + rofs + r) * H_ + nb * 16 + ncol] = acc[nb][r];
}

// ---------------------------------------------------------------- CSR build
__global__ __launch_bounds__(256) void count_deg_kernel(const int* __restrict__ dst,
                                                        int* __restrict__ cnt) {
    int e = blockIdx.x * 256 + threadIdx.x;
    if (e < E_) atomicAdd(&cnt[dst[e]], 1);
}

__global__ __launch_bounds__(1024) void scan_csr_kernel(const int* __restrict__ cnt,
                                                        int* __restrict__ rowptr,
                                                        float* __restrict__ degf,
                                                        float* __restrict__ dis) {
    __shared__ int part[1024];
    const int t = threadIdx.x;
    const int base = t * 32;
    int local[32];
    int sum = 0;
    #pragma unroll
    for (int i = 0; i < 32; ++i) { local[i] = sum; sum += cnt[base + i]; }
    part[t] = sum;
    __syncthreads();
    for (int off = 1; off < 1024; off <<= 1) {
        int v = (t >= off) ? part[t - off] : 0;
        __syncthreads();
        part[t] += v;
        __syncthreads();
    }
    int prefix = (t == 0) ? 0 : part[t - 1];
    #pragma unroll
    for (int i = 0; i < 32; ++i) {
        rowptr[base + i] = prefix + local[i];
        float d = (float)cnt[base + i] + 1.0f;
        degf[base + i] = d;
        dis[base + i]  = rsqrtf(d);
    }
    if (t == 1023) rowptr[N_] = prefix + sum;
}

__global__ __launch_bounds__(256) void fill_csr_kernel(const int* __restrict__ src,
                                                       const int* __restrict__ dst,
                                                       int* __restrict__ cursor,
                                                       int* __restrict__ col) {
    int e = blockIdx.x * 256 + threadIdx.x;
    if (e < E_) {
        int p = atomicAdd(&cursor[dst[e]], 1);
        col[p] = src[e];
    }
}

// ---------------------------------------------------------------- per-node neighbor gather + self loop + bias + relu
// one wave per node; lane owns 4 channels via float4; prefetch next edge's row (global_prefetch_b8)
__global__ __launch_bounds__(256) void gcn_agg_kernel(const float* __restrict__ xw,
                                                      const int* __restrict__ rowptr,
                                                      const int* __restrict__ col,
                                                      const float* __restrict__ dis,
                                                      const float* __restrict__ degf,
                                                      const float* __restrict__ bias,
                                                      float* __restrict__ xout) {
    const int wave = threadIdx.x >> 5;
    const int lane = threadIdx.x & 31;
    const int node = blockIdx.x * 8 + wave;
    const float4* base = (const float4*)xw;

    float ax = 0.f, ay = 0.f, az = 0.f, aw = 0.f;
    const float di = dis[node];
    const int s = rowptr[node], e = rowptr[node + 1];
    for (int j = s; j < e; ++j) {
        const int srcn = col[j];
        if (j + 1 < e)
            __builtin_prefetch(&base[(size_t)col[j + 1] * 32 + lane], 0, 0);
        const float w = di * dis[srcn];
        float4 v = base[(size_t)srcn * 32 + lane];
        ax += w * v.x; ay += w * v.y; az += w * v.z; aw += w * v.w;
    }
    const float inv = 1.0f / degf[node];
    float4 self = base[(size_t)node * 32 + lane];
    ax += inv * self.x; ay += inv * self.y; az += inv * self.z; aw += inv * self.w;
    const float4 bb = ((const float4*)bias)[lane];
    float4 r;
    r.x = fmaxf(ax + bb.x, 0.f);
    r.y = fmaxf(ay + bb.y, 0.f);
    r.z = fmaxf(az + bb.z, 0.f);
    r.w = fmaxf(aw + bb.w, 0.f);
    ((float4*)xout)[(size_t)node * 32 + lane] = r;
}

// ---------------------------------------------------------------- sort-pool: per-graph bitonic top-30 by channel 127
__global__ __launch_bounds__(256) void sortpool_kernel(const float* __restrict__ x,
                                                       float* __restrict__ pooled) {
    __shared__ float key[512];
    __shared__ int   sidx[512];
    const int b = blockIdx.x;
    for (int i = threadIdx.x; i < 512; i += 256) {
        key[i]  = x[(size_t)(b * NPG_ + i) * H_ + (H_ - 1)];
        sidx[i] = i;
    }
    __syncthreads();
    const int t = threadIdx.x;
    for (int k = 2; k <= 512; k <<= 1) {
        for (int j = k >> 1; j > 0; j >>= 1) {
            int i = ((t & ~(j - 1)) << 1) | (t & (j - 1));
            int p = i | j;
            bool dirDesc = ((i & k) == 0);
            float ka = key[i], kb = key[p];
            int   ia = sidx[i], ib = sidx[p];
            bool aFirst = (ka > kb) || (ka == kb && ia < ib);   // desc value, asc index ties
            if (dirDesc ? !aFirst : aFirst) {
                key[i] = kb; key[p] = ka;
                sidx[i] = ib; sidx[p] = ia;
            }
            __syncthreads();
        }
    }
    for (int q = threadIdx.x; q < K_ * H_; q += 256) {
        int kk = q >> 7, h = q & 127;
        pooled[((size_t)b * K_ + kk) * H_ + h] =
            x[(size_t)(b * NPG_ + sidx[kk]) * H_ + h];
    }
}

// ---------------------------------------------------------------- conv1d (NCH, kernel 5, 32 out ch) + relu
__global__ __launch_bounds__(256) void conv1d_kernel(const float* __restrict__ pooled,
                                                     const float* __restrict__ cw,
                                                     const float* __restrict__ cb,
                                                     float* __restrict__ out) {
    __shared__ float pl[K_ * H_];            // 15 KB
    const int b = blockIdx.x;
    for (int t = threadIdx.x; t < K_ * H_; t += 256)
        pl[t] = pooled[(size_t)b * (K_ * H_) + t];
    __syncthreads();
    for (int t = threadIdx.x; t < 32 * (K_ - 4); t += 256) {
        const int o  = t / (K_ - 4);
        const int tt = t % (K_ - 4);
        float s = cb[o];
        for (int kk = 0; kk < 5; ++kk) {
            const float* pr = &pl[(tt + kk) * H_];
            for (int i = 0; i < H_; ++i)
                s += pr[i] * cw[o * (H_ * 5) + i * 5 + kk];
        }
        out[(size_t)b * (32 * (K_ - 4)) + t] = fmaxf(s, 0.f);   // flat index o*(K-4)+tt
    }
}

// ---------------------------------------------------------------- MLP head
__global__ __launch_bounds__(128) void lin1_kernel(const float* __restrict__ sc,
                                                   const float* __restrict__ fc,
                                                   const float* __restrict__ W,
                                                   const float* __restrict__ bvec,
                                                   float* __restrict__ h1) {
    const int b = blockIdx.x, c = threadIdx.x;
    float s = bvec[c];
    for (int k = 0; k < 832; ++k) s += sc[(size_t)b * 832 + k] * W[k * 128 + c];
    for (int k = 0; k < 832; ++k) s += fc[(size_t)b * 832 + k] * W[(832 + k) * 128 + c];
    h1[b * 128 + c] = fmaxf(s, 0.f);
}

__global__ __launch_bounds__(64) void lin2_kernel(const float* __restrict__ h1,
                                                  const float* __restrict__ W,
                                                  const float* __restrict__ bvec,
                                                  float* __restrict__ h2) {
    const int b = blockIdx.x, c = threadIdx.x;
    float s = bvec[c];
    for (int k = 0; k < 128; ++k) s += h1[b * 128 + k] * W[k * 64 + c];
    h2[b * 64 + c] = fmaxf(s, 0.f);
}

__global__ __launch_bounds__(64) void lin3_kernel(const float* __restrict__ h2,
                                                  const float* __restrict__ W,
                                                  const float* __restrict__ bvec,
                                                  float* __restrict__ x3) {
    const int b = blockIdx.x, c = threadIdx.x;
    if (c >= 2) return;
    float s = bvec[c];
    for (int k = 0; k < 64; ++k) s += h2[b * 64 + k] * W[k * 2 + c];
    x3[b * 2 + c] = s;
}

// ---------------------------------------------------------------- CLIP: row norms
__global__ __launch_bounds__(256) void norms_kernel(const float* __restrict__ sc,
                                                    const float* __restrict__ fc,
                                                    float* __restrict__ norms) {
    __shared__ float red[256];
    const int rb = blockIdx.x;                    // 0..63 sc, 64..127 fc
    const float* A = (rb < 64) ? sc : fc;
    const int i = rb & 63;
    float s = 0.f;
    for (int t = threadIdx.x; t < 832; t += 256) {
        float v = A[(size_t)i * 832 + t];
        s += v * v;
    }
    red[threadIdx.x] = s;
    __syncthreads();
    for (int st = 128; st > 0; st >>= 1) {
        if (threadIdx.x < st) red[threadIdx.x] += red[threadIdx.x + st];
        __syncthreads();
    }
    if (threadIdx.x == 0) norms[rb] = sqrtf(red[0]);
}

// ---------------------------------------------------------------- CLIP: per-row logits + logsumexp, accumulate loss
__global__ __launch_bounds__(128) void clip_kernel(const float* __restrict__ sc,
                                                   const float* __restrict__ fc,
                                                   const float* __restrict__ norms,
                                                   float* __restrict__ loss) {
    __shared__ float arow[832];
    __shared__ float lg[128];
    __shared__ float red[128];
    const int rb = blockIdx.x;            // 0..63 sc rows, 64..127 fc rows
    const int isFc = rb >> 6;
    const int i = rb & 63;
    const float* A  = isFc ? fc : sc;
    const float* Bm = isFc ? sc : fc;
    const float* nA = norms + (isFc ? 64 : 0);
    const float* nB = norms + (isFc ? 0 : 64);

    for (int t = threadIdx.x; t < 832; t += 128) arow[t] = A[(size_t)i * 832 + t];
    __syncthreads();

    const int j  = threadIdx.x;
    const int neg = j >> 6;
    const int jj  = j & 63;
    const float* other = neg ? A : Bm;
    float d = 0.f;
    for (int t = 0; t < 832; ++t) d += arow[t] * other[(size_t)jj * 832 + t];
    const float nj = neg ? nA[jj] : nB[jj];
    float l = (d / (nA[i] * nj)) * 2.0f;          // 1/TEMP = 2
    if (neg) l *= 0.8f * ((jj == i) ? 0.f : 1.f); // NEG_W and diag mask (logit forced to 0)
    lg[j] = l;
    red[j] = l;
    __syncthreads();
    for (int s = 64; s > 0; s >>= 1) {
        if (j < s) red[j] = fmaxf(red[j], red[j + s]);
        __syncthreads();
    }
    const float m = red[0];
    __syncthreads();
    red[j] = __expf(l - m);
    __syncthreads();
    for (int s = 64; s > 0; s >>= 1) {
        if (j < s) red[j] += red[j + s];
        __syncthreads();
    }
    if (j == 0) {
        float lse = m + __logf(red[0]);
        atomicAdd(loss, (lse - lg[i]) * (1.0f / 128.0f));
    }
}

// ---------------------------------------------------------------- final: log_softmax(x3) + loss
__global__ __launch_bounds__(64) void final_kernel(const float* __restrict__ x3,
                                                   const float* __restrict__ loss,
                                                   float* __restrict__ out) {
    const int b = threadIdx.x;
    if (b >= 64) return;
    const float a = x3[b * 2], c = x3[b * 2 + 1];
    const float m = fmaxf(a, c);
    const float lse = m + __logf(__expf(a - m) + __expf(c - m));
    const float L = *loss;
    out[b * 2]     = a - lse + L;
    out[b * 2 + 1] = c - lse + L;
}

// ================================================================ host
extern "C" void kernel_launch(void* const* d_in, const int* in_sizes, int n_in,
                              void* d_out, int out_size, void* d_ws, size_t ws_size,
                              hipStream_t stream) {
    (void)in_sizes; (void)n_in; (void)out_size; (void)ws_size;
    char* w = (char*)d_ws;
    size_t off = 0;
    auto carve = [&](size_t bytes) { char* p = w + off; off += (bytes + 255) & ~(size_t)255; return p; };

    float*  xcur   = (float*)carve((size_t)N_ * H_ * 4);
    float*  xw     = (float*)carve((size_t)N_ * H_ * 4);
    __bf16* xbf    = (__bf16*)carve((size_t)N_ * H_ * 2);
    __bf16* wpack  = (__bf16*)carve(16384 * 2);
    int*    cnt    = (int*)carve((size_t)N_ * 4);
    int*    rowptr = (int*)carve((size_t)(N_ + 1) * 4);
    int*    cursor = (int*)carve((size_t)N_ * 4);
    int*    col    = (int*)carve((size_t)E_ * 4);
    float*  degf   = (float*)carve((size_t)N_ * 4);
    float*  dis    = (float*)carve((size_t)N_ * 4);
    float*  pooled = (float*)carve((size_t)B_ * K_ * H_ * 4);
    float*  sc_out = (float*)carve((size_t)B_ * 832 * 4);
    float*  fc_out = (float*)carve((size_t)B_ * 832 * 4);
    float*  norms  = (float*)carve(128 * 4);
    float*  h1     = (float*)carve((size_t)B_ * 128 * 4);
    float*  h2     = (float*)carve((size_t)B_ * 64 * 4);
    float*  x3     = (float*)carve((size_t)B_ * 2 * 4);
    float*  loss   = (float*)carve(4);

    struct Branch {
        const float* x; const int* ei; const float* Ws; const float* bs;
        const float* cw; const float* cb; float* out;
    } br[2] = {
        { (const float*)d_in[0], (const int*)d_in[2], (const float*)d_in[4],
          (const float*)d_in[5], (const float*)d_in[8], (const float*)d_in[9], sc_out },
        { (const float*)d_in[1], (const int*)d_in[3], (const float*)d_in[6],
          (const float*)d_in[7], (const float*)d_in[10], (const float*)d_in[11], fc_out },
    };

    for (int bi = 0; bi < 2; ++bi) {
        const int* src = br[bi].ei;
        const int* dst = br[bi].ei + E_;

        // CSR build (per branch; reused for all 4 layers)
        hipMemsetAsync(cnt, 0, (size_t)N_ * 4, stream);
        count_deg_kernel<<<E_ / 256, 256, 0, stream>>>(dst, cnt);
        scan_csr_kernel<<<1, 1024, 0, stream>>>(cnt, rowptr, degf, dis);
        hipMemcpyAsync(cursor, rowptr, (size_t)N_ * 4, hipMemcpyDeviceToDevice, stream);
        fill_csr_kernel<<<E_ / 256, 256, 0, stream>>>(src, dst, cursor, col);

        for (int l = 0; l < LAYERS_; ++l) {
            const float* xsrc = (l == 0) ? br[bi].x : xcur;
            cvt_bf16_kernel<<<(N_ * H_) / 256, 256, 0, stream>>>(xsrc, xbf, N_ * H_);
            pack_w_kernel<<<64, 256, 0, stream>>>(br[bi].Ws + (size_t)l * H_ * H_, wpack);
            gcn_gemm_kernel<<<N_ / 128, 256, 0, stream>>>(xbf, wpack, xw);
            gcn_agg_kernel<<<N_ / 8, 256, 0, stream>>>(xw, rowptr, col, dis, degf,
                                                       br[bi].bs + (size_t)l * H_, xcur);
        }
        sortpool_kernel<<<B_, 256, 0, stream>>>(xcur, pooled);
        conv1d_kernel<<<B_, 256, 0, stream>>>(pooled, br[bi].cw, br[bi].cb, br[bi].out);
    }

    // MLP head
    lin1_kernel<<<B_, 128, 0, stream>>>(sc_out, fc_out, (const float*)d_in[12],
                                        (const float*)d_in[13], h1);
    lin2_kernel<<<B_, 64, 0, stream>>>(h1, (const float*)d_in[14],
                                       (const float*)d_in[15], h2);
    lin3_kernel<<<B_, 64, 0, stream>>>(h2, (const float*)d_in[16],
                                       (const float*)d_in[17], x3);

    // CLIP loss
    norms_kernel<<<128, 256, 0, stream>>>(sc_out, fc_out, norms);
    hipMemsetAsync(loss, 0, 4, stream);
    clip_kernel<<<128, 128, 0, stream>>>(sc_out, fc_out, norms, loss);

    final_kernel<<<1, 64, 0, stream>>>(x3, loss, (float*)d_out);
}
